// ProtoNetMetric_PT_29480655520056
// MI455X (gfx1250) — matrix operand order repaired and checked by reading, not compiled, not run on hardware
//
#include <hip/hip_runtime.h>

// ProtoNet squared-distance metric:
//   out[n,k] = (alpha+1e-16)^2 * (||x_n||^2 + ||p_k||^2 - 2 * <x_n, p_k>)
// Strategy: split-bf16 (hi/lo) GEMM on v_wmma_f32_16x16x32_bf16 (3 products,
// f32 accumulate -> ~fp32 accuracy at bf16 tensor-op rate), fused epilogue.
// Next K-slab is pulled into WGP$/L2 via global_prefetch (no VGPR cost) while
// the current LDS tile feeds the WMMAs.

typedef __attribute__((ext_vector_type(16))) __bf16 v16bf;
typedef __attribute__((ext_vector_type(8)))  float  v8f;

#define N_ROWS  16384
#define K_PROTO 2048
#define DIM     1024

#define BM  128
#define BN  128
#define BK  32
#define LDK 40   // padded LDS row stride in bf16 elements (80B -> conflict-light)

union FragU { uint4 u[2]; v16bf v; };

// A-matrix 16x32 bf16 fragment (ISA 7.12.2): lanes 0-15 hold K {0..7,16..23},
// lanes 16-31 hold K {8..15,24..31}; rowp points at LDS row (M), koff = 0 or 8.
__device__ __forceinline__ v16bf load_fragA(const __bf16* rowp, int koff) {
  FragU f;
  f.u[0] = *(const uint4*)(rowp + koff);        // K = koff   .. koff+7
  f.u[1] = *(const uint4*)(rowp + koff + 16);   // K = koff+16.. koff+23
  return f.v;
}

// B-matrix 32x16 bf16 fragment: lanes 0-15 hold K {0..7,16..23},
// lanes 16-31 hold K {8..15,24..31}; rowp points at LDS row (N column),
// boff = 0 or 8.
__device__ __forceinline__ v16bf load_fragB(const __bf16* rowp, int boff) {
  FragU f;
  f.u[0] = *(const uint4*)(rowp + boff);        // K = boff   .. boff+7
  f.u[1] = *(const uint4*)(rowp + boff + 16);   // K = boff+16.. boff+23
  return f.v;
}

// -------- Kernel 1: f32 -> (bf16 hi, bf16 lo) planes + row sum-of-squares ----
__global__ __launch_bounds__(128) void split_kernel(
    const float* __restrict__ x, const float* __restrict__ p,
    __bf16* __restrict__ xhi, __bf16* __restrict__ xlo,
    __bf16* __restrict__ phi, __bf16* __restrict__ plo,
    float* __restrict__ xsq, float* __restrict__ psq)
{
  const int row = blockIdx.x;
  const int t   = threadIdx.x;

  const float* src;
  __bf16 *dhi, *dlo;
  float* dsq;
  if (row < N_ROWS) {
    src = x   + (size_t)row * DIM;
    dhi = xhi + (size_t)row * DIM;
    dlo = xlo + (size_t)row * DIM;
    dsq = xsq + row;
  } else {
    const int r = row - N_ROWS;
    src = p   + (size_t)r * DIM;
    dhi = phi + (size_t)r * DIM;
    dlo = plo + (size_t)r * DIM;
    dsq = psq + r;
  }

  const float4 v0 = ((const float4*)src)[t * 2 + 0];
  const float4 v1 = ((const float4*)src)[t * 2 + 1];
  const float vals[8] = {v0.x, v0.y, v0.z, v0.w, v1.x, v1.y, v1.z, v1.w};

  union { __bf16 h[8]; uint4 u; } H, L;
  float s = 0.0f;
#pragma unroll
  for (int i = 0; i < 8; ++i) {
    const float v = vals[i];
    s += v * v;
    const __bf16 h = (__bf16)v;          // hi part
    const float  hf = (float)h;
    H.h[i] = h;
    L.h[i] = (__bf16)(v - hf);           // lo residual
  }
  ((uint4*)dhi)[t] = H.u;
  ((uint4*)dlo)[t] = L.u;

  __shared__ float red[128];
  red[t] = s;
  __syncthreads();
#pragma unroll
  for (int off = 64; off > 0; off >>= 1) {
    if (t < off) red[t] += red[t + off];
    __syncthreads();
  }
  if (t == 0) *dsq = red[0];
}

// -------- Kernel 2: tiled split-bf16 WMMA GEMM with fused distance epilogue --
__global__ __launch_bounds__(256, 1) void protonet_gemm(
    const __bf16* __restrict__ xhi, const __bf16* __restrict__ xlo,
    const __bf16* __restrict__ phi, const __bf16* __restrict__ plo,
    const float* __restrict__ xsq, const float* __restrict__ psq,
    const float* __restrict__ alpha, float* __restrict__ out)
{
  __shared__ __align__(16) __bf16 sXh[BM * LDK];
  __shared__ __align__(16) __bf16 sXl[BM * LDK];
  __shared__ __align__(16) __bf16 sPh[BN * LDK];
  __shared__ __align__(16) __bf16 sPl[BN * LDK];

  const int t     = threadIdx.x;
  const int wave  = t >> 5;
  const int lane  = t & 31;
  const int waveM = wave >> 1;           // 0..3  -> 32-row slab
  const int waveN = wave & 1;            // 0..1  -> 64-col slab
  const int bM = blockIdx.y * BM;
  const int bN = blockIdx.x * BN;

  const int mn    = lane & 15;           // A row / B column within 16x16 tile
  const int hi16  = lane >> 4;
  const int koffA = hi16 * 8;
  const int koffB = hi16 * 8;

  v8f acc[2][4];
#pragma unroll
  for (int i = 0; i < 2; ++i)
#pragma unroll
    for (int j = 0; j < 4; ++j)
      acc[i][j] = (v8f){};

  const int lrow = t >> 2;               // 0..63
  const int lcol = (t & 3) << 3;         // 0,8,16,24

  for (int kk = 0; kk < DIM; kk += BK) {
    // ---- global (L2-resident bf16 planes) -> LDS, 16B per access ----
#pragma unroll
    for (int i = 0; i < 2; ++i) {
      const int r = lrow + i * 64;
      const size_t gx = (size_t)(bM + r) * DIM + kk + lcol;
      const size_t gp = (size_t)(bN + r) * DIM + kk + lcol;
      *(uint4*)&sXh[r * LDK + lcol] = *(const uint4*)&xhi[gx];
      *(uint4*)&sXl[r * LDK + lcol] = *(const uint4*)&xlo[gx];
      *(uint4*)&sPh[r * LDK + lcol] = *(const uint4*)&phi[gp];
      *(uint4*)&sPl[r * LDK + lcol] = *(const uint4*)&plo[gp];
    }

    // ---- prefetch next slab into WGP$/L2 (no VGPR / counter cost); the
    //      fetch completes in the shadow of this iteration's WMMA phase ----
    const int kn = kk + BK;
    if (kn < DIM) {
#pragma unroll
      for (int i = 0; i < 2; ++i) {
        const int r = lrow + i * 64;
        const size_t gx = (size_t)(bM + r) * DIM + kn + lcol;
        const size_t gp = (size_t)(bN + r) * DIM + kn + lcol;
        __builtin_prefetch(&xhi[gx], 0, 3);
        __builtin_prefetch(&xlo[gx], 0, 3);
        __builtin_prefetch(&phi[gp], 0, 3);
        __builtin_prefetch(&plo[gp], 0, 3);
      }
    }
    __syncthreads();

    // ---- fragments ----
    v16bf aH[2], aL[2], bH[4], bL[4];
#pragma unroll
    for (int mt = 0; mt < 2; ++mt) {
      const int row = waveM * 32 + mt * 16 + mn;
      aH[mt] = load_fragA(&sXh[row * LDK], koffA);
      aL[mt] = load_fragA(&sXl[row * LDK], koffA);
    }
#pragma unroll
    for (int nt = 0; nt < 4; ++nt) {
      const int col = waveN * 64 + nt * 16 + mn;
      bH[nt] = load_fragB(&sPh[col * LDK], koffB);
      bL[nt] = load_fragB(&sPl[col * LDK], koffB);
    }

    // ---- cross ~= hi*hi + hi*lo + lo*hi, f32 accumulate ----
#pragma unroll
    for (int mt = 0; mt < 2; ++mt)
#pragma unroll
      for (int nt = 0; nt < 4; ++nt) {
        acc[mt][nt] = __builtin_amdgcn_wmma_f32_16x16x32_bf16(
            false, aH[mt], false, bH[nt], (short)0, acc[mt][nt], false, false);
        acc[mt][nt] = __builtin_amdgcn_wmma_f32_16x16x32_bf16(
            false, aH[mt], false, bL[nt], (short)0, acc[mt][nt], false, false);
        acc[mt][nt] = __builtin_amdgcn_wmma_f32_16x16x32_bf16(
            false, aL[mt], false, bH[nt], (short)0, acc[mt][nt], false, false);
      }
    __syncthreads();
  }

  // ---- epilogue: out = scale * (||x||^2 + ||p||^2 - 2*cross) ----
  const float a0    = alpha[0] + 1e-16f;
  const float scale = a0 * a0;
#pragma unroll
  for (int mt = 0; mt < 2; ++mt) {
    const int rowBase = bM + waveM * 32 + mt * 16 + hi16 * 8;  // C/D: VGPR r -> M=r / r+8
#pragma unroll
    for (int nt = 0; nt < 4; ++nt) {
      const int col = bN + waveN * 64 + nt * 16 + mn;          // C/D: lane -> N
      const float pq = psq[col];
#pragma unroll
      for (int r = 0; r < 8; ++r) {
        const int row = rowBase + r;
        out[(size_t)row * K_PROTO + col] =
            scale * (xsq[row] + pq - 2.0f * acc[mt][nt][r]);
      }
    }
  }
}

extern "C" void kernel_launch(void* const* d_in, const int* in_sizes, int n_in,
                              void* d_out, int out_size, void* d_ws, size_t ws_size,
                              hipStream_t stream) {
  (void)in_sizes; (void)n_in; (void)out_size; (void)ws_size;
  const float* x     = (const float*)d_in[0];
  const float* p     = (const float*)d_in[1];
  const float* alpha = (const float*)d_in[2];
  float* out = (float*)d_out;

  // Workspace layout (bytes): xhi, xlo, phi, plo, xsq, psq  (~75.6 MB)
  char* ws = (char*)d_ws;
  __bf16* xhi = (__bf16*)(ws);
  __bf16* xlo = (__bf16*)(ws + (size_t)N_ROWS * DIM * 2);
  __bf16* phi = (__bf16*)(ws + (size_t)N_ROWS * DIM * 4);
  __bf16* plo = (__bf16*)(ws + (size_t)N_ROWS * DIM * 4 + (size_t)K_PROTO * DIM * 2);
  float*  xsq = (float*)(ws + (size_t)N_ROWS * DIM * 4 + (size_t)K_PROTO * DIM * 4);
  float*  psq = xsq + N_ROWS;

  split_kernel<<<N_ROWS + K_PROTO, 128, 0, stream>>>(x, p, xhi, xlo, phi, plo, xsq, psq);

  dim3 grid(K_PROTO / BN, N_ROWS / BM);   // 16 x 128 = 2048 workgroups
  protonet_gemm<<<grid, 256, 0, stream>>>(xhi, xlo, phi, plo, xsq, psq, alpha, out);
}
